// OneHot_fc_11613591568525
// MI455X (gfx1250) — matrix operand order repaired and checked by reading, not compiled, not run on hardware
//
#include <hip/hip_runtime.h>

#define BATCH 16384
#define HID   1024
#define OUTF  1024

typedef unsigned short u16;
typedef __attribute__((ext_vector_type(16))) __bf16 v16bf;
typedef __attribute__((ext_vector_type(8)))  float  v8f;
typedef __attribute__((ext_vector_type(4)))  float  v4f;   // native vector for NT loads

// ---------- bf16 conversion (round-to-nearest-even) ----------
__device__ __forceinline__ u16 f2bf(float f) {
    unsigned u = __float_as_uint(f);
    unsigned r = u + 0x7FFFu + ((u >> 16) & 1u);
    return (u16)(r >> 16);
}

// ---------- fragment loads in contiguous k order (same order for A and B) ----------
// A (16x32): lane L holds row M=L%16; 16 contiguous K values starting at
//            (L/16)*16.  p = row_base + (L/16)*16.
__device__ __forceinline__ v16bf load_frag_a(const u16* p) {
    union { uint4 u[2]; v16bf v; } f;
    f.u[0] = *(const uint4*)(p);
    f.u[1] = *(const uint4*)(p + 8);
    return f.v;
}
// B (32x16): lane L holds col N=L%16; 16 contiguous K values starting at
//            (L/16)*16.  p = col_row_base + (L/16)*16.
__device__ __forceinline__ v16bf load_frag_b(const u16* p) {
    union { uint4 u[2]; v16bf v; } f;
    f.u[0] = *(const uint4*)(p);
    f.u[1] = *(const uint4*)(p + 8);
    return f.v;
}

// ---------- prep: h = relu(psi1_wT[x] + psi1_b) -> bf16 ----------
// psi1_wT gather rows are used exactly once -> non-temporal loads so the
// 64 MB stream does not evict the L2-resident h/psi2 working set.
__global__ __launch_bounds__(256) void prep_h_kernel(
    const int* __restrict__ x, const float* __restrict__ psi1_wT,
    const float* __restrict__ psi1_b, u16* __restrict__ hbf) {
    size_t idx = (size_t)blockIdx.x * 256 + threadIdx.x;
    int b = (int)(idx >> 7);            // HID/8 = 128 chunks per row
    int j = (int)(idx & 127) << 3;
    int xb = x[b];
    const v4f* w  = (const v4f*)(psi1_wT + (size_t)xb * HID + j);
    const v4f* bs = (const v4f*)(psi1_b + j);
    v4f v0 = __builtin_nontemporal_load(w);
    v4f v1 = __builtin_nontemporal_load(w + 1);
    v4f b0 = bs[0], b1 = bs[1];
    float e[8] = { v0.x + b0.x, v0.y + b0.y, v0.z + b0.z, v0.w + b0.w,
                   v1.x + b1.x, v1.y + b1.y, v1.z + b1.z, v1.w + b1.w };
#pragma unroll
    for (int i = 0; i < 8; ++i) e[i] = e[i] > 0.f ? e[i] : 0.f;
    uint4 o;
    o.x = (unsigned)f2bf(e[0]) | ((unsigned)f2bf(e[1]) << 16);
    o.y = (unsigned)f2bf(e[2]) | ((unsigned)f2bf(e[3]) << 16);
    o.z = (unsigned)f2bf(e[4]) | ((unsigned)f2bf(e[5]) << 16);
    o.w = (unsigned)f2bf(e[6]) | ((unsigned)f2bf(e[7]) << 16);
    *(uint4*)(hbf + idx * 8) = o;    // regular store: keep h in L2
}

// ---------- prep: psi2_w f32 -> bf16 ----------
__global__ __launch_bounds__(256) void prep_w_kernel(
    const float* __restrict__ psi2_w, u16* __restrict__ wbf) {
    size_t idx = (size_t)blockIdx.x * 256 + threadIdx.x;   // OUTF*HID/8 threads
    const v4f* w = (const v4f*)(psi2_w + idx * 8);
    v4f v0 = w[0], v1 = w[1];
    uint4 o;
    o.x = (unsigned)f2bf(v0.x) | ((unsigned)f2bf(v0.y) << 16);
    o.y = (unsigned)f2bf(v0.z) | ((unsigned)f2bf(v0.w) << 16);
    o.z = (unsigned)f2bf(v1.x) | ((unsigned)f2bf(v1.y) << 16);
    o.w = (unsigned)f2bf(v1.z) | ((unsigned)f2bf(v1.w) << 16);
    *(uint4*)(wbf + idx * 8) = o;
}

// ---------- GEMM: out = h @ psi2_w^T + psi2_b + wr_T[x] + wr_b ----------
// Block: 128(M) x 128(N), 8 wave32s as 4(M) x 2(N); wave = 32x64 = 2x4 tiles.
__global__ __launch_bounds__(256) void gemm_kernel(
    const u16* __restrict__ hbf, const u16* __restrict__ wbf,
    const float* __restrict__ psi2_b, const int* __restrict__ x,
    const float* __restrict__ wr_T, const float* __restrict__ wr_b,
    float* __restrict__ out) {
    const int lane = threadIdx.x & 31;
    const int wave = threadIdx.x >> 5;
    const int wm = wave & 3;       // 0..3  (M direction)
    const int wn = wave >> 2;      // 0..1  (N direction)
    const int bm = blockIdx.x * 128;
    const int bn = blockIdx.y * 128;

    const int lr    = lane & 15;
    const int hi    = lane >> 4;
    const int aKoff = hi * 16;     // A layout chunk start
    const int bKoff = hi * 16;     // B layout chunk start

    const u16* aptr0 = hbf + (size_t)(bm + wm * 32 + lr) * HID + aKoff;
    const u16* aptr1 = aptr0 + (size_t)16 * HID;
    const u16* bptr[4];
#pragma unroll
    for (int nt = 0; nt < 4; ++nt)
        bptr[nt] = wbf + (size_t)(bn + wn * 64 + nt * 16 + lr) * HID + bKoff;

    v8f acc[2][4] = {};

    // Unroll x2: two independent fragment register sets let the scheduler
    // fill WMMA->VALU WAR hazard slots with the next step's loads/address math.
#pragma unroll 2
    for (int k = 0; k < HID; k += 32) {
        v16bf a0 = load_frag_a(aptr0 + k);
        v16bf a1 = load_frag_a(aptr1 + k);
        v16bf bf[4];
#pragma unroll
        for (int nt = 0; nt < 4; ++nt) bf[nt] = load_frag_b(bptr[nt] + k);
#pragma unroll
        for (int nt = 0; nt < 4; ++nt) {
            acc[0][nt] = __builtin_amdgcn_wmma_f32_16x16x32_bf16(
                false, a0, false, bf[nt], (short)0, acc[0][nt], false, false);
            acc[1][nt] = __builtin_amdgcn_wmma_f32_16x16x32_bf16(
                false, a1, false, bf[nt], (short)0, acc[1][nt], false, false);
        }
    }

    // Epilogue: C/D tile layout — element i: M = i + 8*(lane>=16), N = lane%16.
    // wr_T gather rows + output are streamed once -> non-temporal to keep
    // h_bf16 / psi2_bf16 resident in the 192 MB L2.
#pragma unroll
    for (int mt = 0; mt < 2; ++mt) {
        const int row_base = bm + wm * 32 + mt * 16 + (hi << 3);
        int xrow[8];
#pragma unroll
        for (int i = 0; i < 8; ++i) xrow[i] = x[row_base + i];
#pragma unroll
        for (int nt = 0; nt < 4; ++nt) {
            const int col = bn + wn * 64 + nt * 16 + lr;
            const float cb = psi2_b[col] + wr_b[col];
#pragma unroll
            for (int i = 0; i < 8; ++i) {
                const int row = row_base + i;
                const float wr =
                    __builtin_nontemporal_load(wr_T + (size_t)xrow[i] * OUTF + col);
                __builtin_nontemporal_store(acc[mt][nt][i] + cb + wr,
                                            out + (size_t)row * OUTF + col);
            }
        }
    }
}

extern "C" void kernel_launch(void* const* d_in, const int* in_sizes, int n_in,
                              void* d_out, int out_size, void* d_ws, size_t ws_size,
                              hipStream_t stream) {
    const int*   x       = (const int*)d_in[0];
    const float* psi1_wT = (const float*)d_in[1];
    const float* psi1_b  = (const float*)d_in[2];
    const float* psi2_w  = (const float*)d_in[3];
    const float* psi2_b  = (const float*)d_in[4];
    const float* wr_T    = (const float*)d_in[5];
    const float* wr_b    = (const float*)d_in[6];
    float* out = (float*)d_out;

    u16* hbf = (u16*)d_ws;                                        // 32 MB
    u16* wbf = (u16*)((char*)d_ws + (size_t)BATCH * HID * 2);     // +2 MB

    prep_h_kernel<<<(BATCH * (HID / 8)) / 256, 256, 0, stream>>>(x, psi1_wT, psi1_b, hbf);
    prep_w_kernel<<<(OUTF * (HID / 8)) / 256, 256, 0, stream>>>(psi2_w, wbf);

    dim3 grid(BATCH / 128, OUTF / 128);
    gemm_kernel<<<grid, 256, 0, stream>>>(hbf, wbf, psi2_b, x, wr_T, wr_b, out);
}
